// Prototype_83348135346517
// MI455X (gfx1250) — compile-verified, matrix operand order checked by reading, not run on hardware
//
#include <hip/hip_runtime.h>
#include <hip/hip_bf16.h>

// ---------------------------------------------------------------------------
// out[b,p] = -(||x_b||^2 + ||p_p||^2 - 2 * <x_b, p_p>)
// B=16384, P=1024, H=512, all f32.
// Single fused kernel:
//   - bf16 hi/lo split GEMM on v_wmma_f32_16x16x32_bf16 (3 WMMAs / K-tile)
//   - global->reg->convert->LDS staging, double-buffered
//   - row norms ||x||^2, ||p||^2 accumulated during staging (free, co-executes
//     under WMMA), wave-reduced via shuffles, parked in LDS for the epilogue
// ---------------------------------------------------------------------------

typedef __attribute__((ext_vector_type(16))) __bf16 v16bf;
typedef __attribute__((ext_vector_type(8)))  __bf16 v8bf;
typedef __attribute__((ext_vector_type(8)))  float  v8f;

constexpr int Mdim = 16384;   // BATCH
constexpr int Ndim = 1024;    // N_PROTO
constexpr int Kdim = 512;     // HIDDEN
constexpr int BM = 128;
constexpr int BN = 128;
constexpr int BK = 32;
constexpr int LDT = BK + 8;   // padded LDS row stride (bf16 elems): 80B, 16B-aligned

// split f32x4 into bf16 hi + bf16 lo, pack and store 8B each into LDS
__device__ __forceinline__ void split4_store(const float4& f, __bf16* hp, __bf16* lp) {
  float v[4] = {f.x, f.y, f.z, f.w};
  union U { __bf16 b[4]; uint2 u; } H, L;
#pragma unroll
  for (int j = 0; j < 4; ++j) {
    __bf16 h = (__bf16)v[j];
    H.b[j] = h;
    L.b[j] = (__bf16)(v[j] - (float)h);
  }
  *(uint2*)hp = H.u;
  *(uint2*)lp = L.u;
}

// Load one 16x32 bf16 fragment from an LDS tile (row-major, stride LDT).
// Per ISA 16-bit A layout: lanes 0-15 -> M=lane, K in {0..7, 16..23};
// lanes 16-31 -> M=lane-16, K in {8..15, 24..31}. Two ds_load_b128 per lane.
__device__ __forceinline__ v16bf load_frag(const __bf16* base, int row16, int lane) {
  const int r  = row16 + (lane & 15);
  const int kb = (lane >> 4) << 3;                // 0 or 8
  const __bf16* p = base + r * LDT + kb;
  union { v16bf v; v8bf h[2]; } u;
  u.h[0] = *(const v8bf*)(p);                     // K = kb .. kb+7
  u.h[1] = *(const v8bf*)(p + 16);                // K = 16+kb .. 16+kb+7
  return u.v;
}

__device__ __forceinline__ v8f wmma_bf16(v16bf a, v16bf b, v8f c) {
  return __builtin_amdgcn_wmma_f32_16x16x32_bf16(
      /*neg_a=*/false, a, /*neg_b=*/false, b,
      /*c_mod=*/(short)0, c, /*reuse_a=*/false, /*reuse_b=*/false);
}

// ---------------- fused GEMM + row norms + epilogue ------------------------
__global__ __launch_bounds__(256, 1) void proto_dist_kernel(
    const float* __restrict__ X, const float* __restrict__ P,
    float* __restrict__ out) {
  __shared__ __align__(16) __bf16 sA[2][2][BM * LDT];   // [buf][hi/lo]
  __shared__ __align__(16) __bf16 sB[2][2][BN * LDT];   // B stored as P rows: [N][K]
  __shared__ float sXsq[BM];                            // block-local ||x_row||^2
  __shared__ float sPsq[BN];                            // block-local ||p_row||^2

  const int tid   = threadIdx.x;
  const int lane  = tid & 31;
  const int wave  = tid >> 5;
  const int blkM  = blockIdx.y * BM;
  const int blkN  = blockIdx.x * BN;
  const int waveM = (wave & 3) * 32;                    // 4 waves along M
  const int waveN = (wave >> 2) * 64;                   // 2 waves along N

  v8f acc[2][4] = {};

  float4 ra[4], rb[4];                                  // staging regs
  float  sumA[4] = {0.f, 0.f, 0.f, 0.f};                // sum-of-squares partials
  float  sumB[4] = {0.f, 0.f, 0.f, 0.f};                // (fixed 4 rows per thread)

  auto gload = [&](int k0) {
#pragma unroll
    for (int i = 0; i < 4; ++i) {
      const int ch  = tid + i * 256;                    // 1024 float4 chunks / tile
      const int row = ch >> 3;
      const int col = (ch & 7) << 2;
      ra[i] = *(const float4*)(X + (size_t)(blkM + row) * Kdim + k0 + col);
      rb[i] = *(const float4*)(P + (size_t)(blkN + row) * Kdim + k0 + col);
      sumA[i] += ra[i].x * ra[i].x + ra[i].y * ra[i].y +
                 ra[i].z * ra[i].z + ra[i].w * ra[i].w;
      sumB[i] += rb[i].x * rb[i].x + rb[i].y * rb[i].y +
                 rb[i].z * rb[i].z + rb[i].w * rb[i].w;
    }
  };
  auto lstore = [&](int buf) {
#pragma unroll
    for (int i = 0; i < 4; ++i) {
      const int ch  = tid + i * 256;
      const int row = ch >> 3;
      const int col = (ch & 7) << 2;
      split4_store(ra[i], &sA[buf][0][row * LDT + col], &sA[buf][1][row * LDT + col]);
      split4_store(rb[i], &sB[buf][0][row * LDT + col], &sB[buf][1][row * LDT + col]);
    }
  };

  gload(0);
  lstore(0);
  __syncthreads();

  constexpr int KSTEPS = Kdim / BK;                     // 16
  for (int ks = 0; ks < KSTEPS; ++ks) {
    const int cur = ks & 1;

    if (ks + 1 < KSTEPS) gload((ks + 1) * BK);
    if (ks + 2 < KSTEPS) {                              // global_prefetch_b8 (L2)
      __builtin_prefetch(X + (size_t)(blkM + (tid >> 1)) * Kdim + (ks + 2) * BK, 0, 1);
      __builtin_prefetch(P + (size_t)(blkN + (tid >> 1)) * Kdim + (ks + 2) * BK, 0, 1);
    }

    v16bf ahi[2], alo[2], bhi[4], blo[4];
#pragma unroll
    for (int m = 0; m < 2; ++m) {
      ahi[m] = load_frag(sA[cur][0], waveM + m * 16, lane);
      alo[m] = load_frag(sA[cur][1], waveM + m * 16, lane);
    }
#pragma unroll
    for (int n = 0; n < 4; ++n) {
      bhi[n] = load_frag(sB[cur][0], waveN + n * 16, lane);
      blo[n] = load_frag(sB[cur][1], waveN + n * 16, lane);
    }

#pragma unroll
    for (int m = 0; m < 2; ++m)
#pragma unroll
      for (int n = 0; n < 4; ++n) {
        acc[m][n] = wmma_bf16(ahi[m], bhi[n], acc[m][n]);   // hi*hi
        acc[m][n] = wmma_bf16(ahi[m], blo[n], acc[m][n]);   // hi*lo
        acc[m][n] = wmma_bf16(alo[m], bhi[n], acc[m][n]);   // lo*hi
      }

    if (ks + 1 < KSTEPS) lstore(cur ^ 1);
    __syncthreads();
  }

  // Row-norm reduction: 8 consecutive threads (aligned group within a wave)
  // hold the K-partials of one row; fold with xor-shuffles, park in LDS.
#pragma unroll
  for (int i = 0; i < 4; ++i) {
    float a = sumA[i];
    float b = sumB[i];
#pragma unroll
    for (int off = 4; off > 0; off >>= 1) {
      a += __shfl_xor(a, off, 32);
      b += __shfl_xor(b, off, 32);
    }
    if ((tid & 7) == 0) {
      const int row = (tid >> 3) + 32 * i;              // 0..127
      sXsq[row] = a;
      sPsq[row] = b;
    }
  }
  __syncthreads();

  // Epilogue: out = 2*cross - ||x||^2 - ||p||^2
  // C/D layout: VGPR r -> lanes 0-15: (M=r, N=lane); lanes 16-31: (M=8+r, N=lane-16)
#pragma unroll
  for (int n = 0; n < 4; ++n) {
    const int col_l = waveN + n * 16 + (lane & 15);
    const float pq  = sPsq[col_l];
    const int col   = blkN + col_l;
#pragma unroll
    for (int m = 0; m < 2; ++m) {
      const int rbase = waveM + m * 16 + ((lane >> 4) << 3);
#pragma unroll
      for (int r = 0; r < 8; ++r) {
        const int row_l = rbase + r;
        out[(size_t)(blkM + row_l) * Ndim + col] =
            2.0f * acc[m][n][r] - sXsq[row_l] - pq;
      }
    }
  }
}

extern "C" void kernel_launch(void* const* d_in, const int* in_sizes, int n_in,
                              void* d_out, int out_size, void* d_ws, size_t ws_size,
                              hipStream_t stream) {
  const float* X = (const float*)d_in[0];   // [16384, 512]
  const float* P = (const float*)d_in[1];   // [1024, 512]
  float* out = (float*)d_out;               // [16384, 1024]
  (void)d_ws; (void)ws_size; (void)in_sizes; (void)n_in; (void)out_size;

  dim3 grid(Ndim / BN, Mdim / BM);          // (8, 128)
  proto_dist_kernel<<<grid, 256, 0, stream>>>(X, P, out);
}
